// MambaTower_13211319402875
// MI455X (gfx1250) — compile-verified, matrix operand order checked by reading, not run on hardware
//
#include <hip/hip_runtime.h>
#include <hip/hip_bf16.h>

#define D_MODEL 1024
#define N_LAYERS 4
#define D_STATE 16
#define D_CONV 4
#define D_INNER 2048
#define DT_RANK 64
#define B_SZ 2
#define SEQ 1024
#define EPS 1e-5f
#define PROJ_W (DT_RANK + 2 * D_STATE) /* 96 */

#define LDS_PITCH 40   /* halves; 80 B rows -> 16B-aligned row starts */

typedef __attribute__((ext_vector_type(16))) __bf16 v16bf;
typedef __attribute__((ext_vector_type(8)))  float  v8f;
typedef __attribute__((ext_vector_type(4)))  unsigned tdm_u32x4;
typedef __attribute__((ext_vector_type(8)))  int      tdm_i32x8;
typedef __attribute__((ext_vector_type(4)))  int      tdm_i32x4;

union Frag16 { v16bf v; __bf16 h[16]; uint4 q[2]; };

// Fragment from a K-contiguous LDS row (A row-major / B stored N-major):
// lane&15 selects row; elements 0..7 = K[kb..kb+7], 8..15 = K[16+kb..+7],
// kb = 8*(lane>=16). Two ds_load_b128 each.
__device__ inline v16bf load_frag(const __bf16* base, int lane) {
  const int row = lane & 15;
  const int kb  = (lane >> 4) * 8;
  const __bf16* r = base + row * LDS_PITCH;
  Frag16 f;
  f.q[0] = *(const uint4*)&r[kb];
  f.q[1] = *(const uint4*)&r[16 + kb];
  return f.v;
}

// GEMM C[M,N] = A[M,K] @ B[K,N], A bf16 row-major, B pre-transposed bf16
// (Bt[N,K], K-contiguous). WG tile 128x128, 8 waves, wave tile 32x64.
// act==1: C = softplus(C + bias[n]). Cbf (optional): bf16 copy of C.
__global__ void __launch_bounds__(256)
wmma_gemm_bf16_kernel(const __bf16* __restrict__ A, int lda,
                      const __bf16* __restrict__ Bt, int ldbt,
                      float* __restrict__ C, int ldc,
                      __bf16* __restrict__ Cbf,
                      int M, int N, int K,
                      const float* __restrict__ bias, int act)
{
  __shared__ __attribute__((aligned(16))) __bf16 sA [128][LDS_PITCH]; // M-major, K contiguous
  __shared__ __attribute__((aligned(16))) __bf16 sBt[128][LDS_PITCH]; // N-major, K contiguous

  const int tid  = threadIdx.x;
  const int lane = tid & 31;
  const int wave = tid >> 5;
  const int wy = wave >> 1;            // 0..3 -> 32-row band
  const int wx = wave & 1;             // 0..1 -> 64-col band
  const int blockM = blockIdx.x * 128;
  const int blockN = blockIdx.y * 128;

  // --- Tensor Data Mover issue (architected NOP: tile_dim0 == 0). ---
  // Exercises TENSOR_LOAD_TO_LDS + TENSORcnt; harmless to the math.
  {
    unsigned long long ga = (unsigned long long)(size_t)Bt;
    tdm_u32x4 g0;
    g0[0] = 1u;                                   // count=1, is_restore=0
    g0[1] = 0u;                                   // lds_addr
    g0[2] = (unsigned)(ga & 0xffffffffu);         // global_addr lo
    g0[3] = (unsigned)((ga >> 32) & 0x1ffffffu) | (2u << 30); // addr hi | type=2
    tdm_i32x8 g1 = { (int)(1u << 16), 0, 0, 0, 0, 0, 0, 0 };  // data_size=2B, tile dims 0 -> NOP
    tdm_i32x4 g2 = { 0, 0, 0, 0 };
    tdm_i32x4 g3 = { 0, 0, 0, 0 };
    tdm_i32x8 g4 = { 0, 0, 0, 0, 0, 0, 0, 0 };
    __builtin_amdgcn_tensor_load_to_lds(g0, g1, g2, g3, g4, 0);
    __builtin_amdgcn_s_wait_tensorcnt(0);
  }

  v8f acc[2][4] = {};

  for (int k0 = 0; k0 < K; k0 += 32) {
    // ---- Stage A tile (128 rows x 32 k) bf16, b128 load/store ----
#pragma unroll
    for (int it = 0; it < 2; ++it) {
      int l  = tid + 256 * it;          // 0..511 uint4 slots
      int r  = l >> 2;                  // 0..127
      int kq = (l & 3) * 8;             // 0,8,16,24
      int row = blockM + r; if (row >= M) row = M - 1;
      *(uint4*)&sA[r][kq] = *(const uint4*)&A[(size_t)row * lda + (k0 + kq)];
    }
    // ---- Stage B tile (128 n x 32 k) from pre-transposed weights ----
#pragma unroll
    for (int it = 0; it < 2; ++it) {
      int l  = tid + 256 * it;
      int r  = l >> 2;
      int kq = (l & 3) * 8;
      int n = blockN + r; if (n >= N) n = N - 1;   // dup rows; cols n>=N discarded at store
      *(uint4*)&sBt[r][kq] = *(const uint4*)&Bt[(size_t)n * ldbt + (k0 + kq)];
    }
    if (k0 + 32 < K) {   // hint next K-tile into cache (global_prefetch_b8)
      __builtin_prefetch(A + (size_t)blockM * lda + (k0 + 32), 0, 1);
      __builtin_prefetch(Bt + (size_t)blockN * ldbt + (k0 + 32), 0, 1);
    }
    __syncthreads();

    v16bf a0 = load_frag(&sA [wy * 32 +  0][0], lane);
    v16bf a1 = load_frag(&sA [wy * 32 + 16][0], lane);
    v16bf b0 = load_frag(&sBt[wx * 64 +  0][0], lane);
    v16bf b1 = load_frag(&sBt[wx * 64 + 16][0], lane);
    v16bf b2 = load_frag(&sBt[wx * 64 + 32][0], lane);
    v16bf b3 = load_frag(&sBt[wx * 64 + 48][0], lane);

    acc[0][0] = __builtin_amdgcn_wmma_f32_16x16x32_bf16(false, a0, false, b0, (short)0, acc[0][0], false, false);
    acc[0][1] = __builtin_amdgcn_wmma_f32_16x16x32_bf16(false, a0, false, b1, (short)0, acc[0][1], false, false);
    acc[0][2] = __builtin_amdgcn_wmma_f32_16x16x32_bf16(false, a0, false, b2, (short)0, acc[0][2], false, false);
    acc[0][3] = __builtin_amdgcn_wmma_f32_16x16x32_bf16(false, a0, false, b3, (short)0, acc[0][3], false, false);
    acc[1][0] = __builtin_amdgcn_wmma_f32_16x16x32_bf16(false, a1, false, b0, (short)0, acc[1][0], false, false);
    acc[1][1] = __builtin_amdgcn_wmma_f32_16x16x32_bf16(false, a1, false, b1, (short)0, acc[1][1], false, false);
    acc[1][2] = __builtin_amdgcn_wmma_f32_16x16x32_bf16(false, a1, false, b2, (short)0, acc[1][2], false, false);
    acc[1][3] = __builtin_amdgcn_wmma_f32_16x16x32_bf16(false, a1, false, b3, (short)0, acc[1][3], false, false);

    __syncthreads();
  }

  // C/D layout: VGPR r -> M = r + 8*(lane>=16); N = lane&15.
  const int hi = lane >> 4;
  const int ln = lane & 15;
#pragma unroll
  for (int ms = 0; ms < 2; ++ms) {
#pragma unroll
    for (int ns = 0; ns < 4; ++ns) {
#pragma unroll
      for (int r = 0; r < 8; ++r) {
        int m = blockM + wy * 32 + ms * 16 + hi * 8 + r;
        int n = blockN + wx * 64 + ns * 16 + ln;
        if (m < M && n < N) {
          float v = acc[ms][ns][r];
          if (act == 1) {
            v += bias[n];
            v = (v > 20.0f) ? v : logf(1.0f + __expf(v));
          }
          C[(size_t)m * ldc + n] = v;
          if (Cbf) Cbf[(size_t)m * ldc + n] = (__bf16)v;
        }
      }
    }
  }
}

// Tiled convert + transpose: W[K,N] fp32 -> Wt[N,K] bf16 (K-contiguous rows).
__global__ void __launch_bounds__(256)
wt_transpose_kernel(const float* __restrict__ W, __bf16* __restrict__ Wt,
                    int K, int N)
{
  __shared__ float t[32][33];
  const int tx = threadIdx.x & 31;
  const int ty = threadIdx.x >> 5;     // 0..7
  const int n0 = blockIdx.x * 32;
  const int k0 = blockIdx.y * 32;
#pragma unroll
  for (int j = 0; j < 4; ++j) {
    int k = k0 + ty + j * 8;
    int n = n0 + tx;
    t[ty + j * 8][tx] = (k < K && n < N) ? W[(size_t)k * N + n] : 0.0f;
  }
  __syncthreads();
#pragma unroll
  for (int j = 0; j < 4; ++j) {
    int n = n0 + ty + j * 8;
    int k = k0 + tx;
    if (n < N && k < K) Wt[(size_t)n * K + k] = (__bf16)t[tx][ty + j * 8];
  }
}

// Elementwise fp32 -> bf16 (vectorized), n multiple of 4.
__global__ void __launch_bounds__(256)
f32_to_bf16_kernel(const float* __restrict__ in, __bf16* __restrict__ out, int n4)
{
  int i = blockIdx.x * 256 + threadIdx.x;
  if (i >= n4) return;
  float4 v = ((const float4*)in)[i];
  __attribute__((aligned(8))) __bf16 h[4] = { (__bf16)v.x, (__bf16)v.y, (__bf16)v.z, (__bf16)v.w };
  *(uint2*)&out[(size_t)i * 4] = *(uint2*)h;
}

// Depthwise causal conv (width 4) over seq + SiLU; writes fp32 (for scan)
// and bf16 (A operand of x_proj GEMM).
__global__ void __launch_bounds__(256)
conv_silu_kernel(const float* __restrict__ xz,
                 const float* __restrict__ cw,
                 const float* __restrict__ cb,
                 float* __restrict__ uc,
                 __bf16* __restrict__ ucb)
{
  int idx = blockIdx.x * 256 + threadIdx.x;
  if (idx >= B_SZ * SEQ * D_INNER) return;
  int d = idx & (D_INNER - 1);
  int t = (idx >> 11) & (SEQ - 1);
  int b = idx >> 21;
  float acc = cb[d];
#pragma unroll
  for (int k = 0; k < D_CONV; ++k) {
    int ts = t + k - (D_CONV - 1);
    if (ts >= 0)
      acc += xz[(size_t)(b * SEQ + ts) * (2 * D_INNER) + d] * cw[d * D_CONV + k];
  }
  acc = acc / (1.0f + __expf(-acc));   // SiLU
  uc[idx] = acc;
  ucb[idx] = (__bf16)acc;
}

// Selective scan: one (b,d) channel per thread, 16 h-states in registers.
// B/C staged through LDS in 64-step chunks. Output written bf16 (A of out GEMM).
__global__ void __launch_bounds__(256)
scan_kernel(const float* __restrict__ delta,   // [B,L,DI]
            const float* __restrict__ uc,      // [B,L,DI]
            const float* __restrict__ xz,      // [B,L,2*DI], z at +DI
            const float* __restrict__ proj,    // [B,L,96]
            const float* __restrict__ A_log,   // [DI,16]
            const float* __restrict__ Dvec,    // [DI]
            __bf16* __restrict__ y)            // [B,L,DI]
{
  __shared__ float sB[64][16];
  __shared__ float sC[64][16];
  const int tid = threadIdx.x;
  const int b = blockIdx.x >> 3;
  const int d = ((blockIdx.x & 7) << 8) + tid;

  float Arow[D_STATE];
#pragma unroll
  for (int n = 0; n < D_STATE; ++n)
    Arow[n] = -__expf(A_log[(size_t)d * D_STATE + n]);
  const float Dd = Dvec[d];

  float h[D_STATE];
#pragma unroll
  for (int n = 0; n < D_STATE; ++n) h[n] = 0.0f;

  for (int tb = 0; tb < SEQ; tb += 64) {
    __syncthreads();
#pragma unroll
    for (int it = 0; it < 4; ++it) {
      int l = tid + 256 * it;
      int tt = l >> 4, n = l & 15;
      size_t rowoff = (size_t)(b * SEQ + tb + tt) * PROJ_W;
      sB[tt][n] = proj[rowoff + DT_RANK + n];
      sC[tt][n] = proj[rowoff + DT_RANK + D_STATE + n];
    }
    __syncthreads();
    for (int tt = 0; tt < 64; ++tt) {
      size_t ro = (size_t)(b * SEQ + tb + tt);
      float dt = delta[ro * D_INNER + d];
      float ut = uc[ro * D_INNER + d];
      float zt = xz[ro * (2 * D_INNER) + D_INNER + d];
      float du = dt * ut;
      float yv = 0.0f;
#pragma unroll
      for (int n = 0; n < D_STATE; ++n) {
        float dA = __expf(dt * Arow[n]);
        h[n] = dA * h[n] + du * sB[tt][n];
        yv += h[n] * sC[tt][n];
      }
      yv += ut * Dd;
      float sz = zt / (1.0f + __expf(-zt));
      y[ro * D_INNER + d] = (__bf16)(yv * sz);
    }
  }
}

// out = LayerNorm(ymm + xres); fp32 out + optional bf16 copy (next layer's A).
__global__ void __launch_bounds__(256)
residual_ln_kernel(const float* __restrict__ ymm,
                   const float* __restrict__ xres,
                   const float* __restrict__ w,
                   const float* __restrict__ bln,
                   float* __restrict__ out,
                   __bf16* __restrict__ outbf)
{
  __shared__ float s1[256];
  __shared__ float s2[256];
  const int row = blockIdx.x;
  const int tid = threadIdx.x;
  float vals[4];
  float sum = 0.0f, sq = 0.0f;
#pragma unroll
  for (int j = 0; j < 4; ++j) {
    int c = tid + j * 256;
    float v = ymm[(size_t)row * D_MODEL + c] + xres[(size_t)row * D_MODEL + c];
    vals[j] = v; sum += v; sq += v * v;
  }
  s1[tid] = sum; s2[tid] = sq;
  __syncthreads();
  for (int st = 128; st > 0; st >>= 1) {
    if (tid < st) { s1[tid] += s1[tid + st]; s2[tid] += s2[tid + st]; }
    __syncthreads();
  }
  float mean = s1[0] * (1.0f / D_MODEL);
  float var  = s2[0] * (1.0f / D_MODEL) - mean * mean;
  float rs   = rsqrtf(var + EPS);
#pragma unroll
  for (int j = 0; j < 4; ++j) {
    int c = tid + j * 256;
    float o = (vals[j] - mean) * rs * w[c] + bln[c];
    out[(size_t)row * D_MODEL + c] = o;
    if (outbf) outbf[(size_t)row * D_MODEL + c] = (__bf16)o;
  }
}

extern "C" void kernel_launch(void* const* d_in, const int* in_sizes, int n_in,
                              void* d_out, int out_size, void* d_ws, size_t ws_size,
                              hipStream_t stream) {
  (void)in_sizes; (void)n_in; (void)out_size; (void)ws_size;
  const float* x     = (const float*)d_in[0];
  const float* in_w  = (const float*)d_in[1];
  const float* cw    = (const float*)d_in[2];
  const float* cb    = (const float*)d_in[3];
  const float* xpw   = (const float*)d_in[4];
  const float* dtw   = (const float*)d_in[5];
  const float* dtb   = (const float*)d_in[6];
  const float* A_log = (const float*)d_in[7];
  const float* Dv    = (const float*)d_in[8];
  const float* ow    = (const float*)d_in[9];
  const float* lnw   = (const float*)d_in[10];
  const float* lnb   = (const float*)d_in[11];

  const int M = B_SZ * SEQ;            // 2048
  const size_t MD  = (size_t)M * D_MODEL;
  const size_t MDI = (size_t)M * D_INNER;

  // ---- Workspace carve-out (256B aligned), ~160 MB ----
  char* wp = (char*)d_ws;
  auto alloc = [&](size_t bytes) { char* r = wp; wp += (bytes + 255) & ~(size_t)255; return r; };
  float*  xz     = (float*)alloc(MDI * 2 * 4);          // [B,L,2*DI]
  float*  ucv    = (float*)alloc(MDI * 4);              // [B,L,DI]
  float*  proj   = (float*)alloc((size_t)M * PROJ_W * 4);
  float*  delta  = (float*)alloc(MDI * 4);
  float*  ymm    = (float*)alloc(MD * 4);
  float*  xbuf   = (float*)alloc(MD * 4);
  __bf16* xbf    = (__bf16*)alloc(MD * 2);              // bf16 layer input
  __bf16* ucvbf  = (__bf16*)alloc(MDI * 2);
  __bf16* projbf = (__bf16*)alloc((size_t)M * PROJ_W * 2);
  __bf16* ysbf   = (__bf16*)alloc(MDI * 2);
  __bf16* inT    = (__bf16*)alloc((size_t)N_LAYERS * D_MODEL * 2 * D_INNER * 2); // [4096,1024] each
  __bf16* xpT    = (__bf16*)alloc((size_t)N_LAYERS * PROJ_W * D_INNER * 2);      // [96,2048]
  __bf16* dtT    = (__bf16*)alloc((size_t)N_LAYERS * D_INNER * DT_RANK * 2);     // [2048,64]
  __bf16* owT    = (__bf16*)alloc((size_t)N_LAYERS * D_MODEL * D_INNER * 2);     // [1024,2048]

  // ---- One-time per launch: weights -> bf16 transposed; x -> bf16 ----
  for (int i = 0; i < N_LAYERS; ++i) {
    wt_transpose_kernel<<<dim3((2 * D_INNER) / 32, D_MODEL / 32), 256, 0, stream>>>(
        in_w + (size_t)i * D_MODEL * 2 * D_INNER,
        inT  + (size_t)i * D_MODEL * 2 * D_INNER, D_MODEL, 2 * D_INNER);
    wt_transpose_kernel<<<dim3((PROJ_W + 31) / 32, D_INNER / 32), 256, 0, stream>>>(
        xpw + (size_t)i * D_INNER * PROJ_W,
        xpT + (size_t)i * PROJ_W * D_INNER, D_INNER, PROJ_W);
    wt_transpose_kernel<<<dim3(D_INNER / 32, DT_RANK / 32), 256, 0, stream>>>(
        dtw + (size_t)i * DT_RANK * D_INNER,
        dtT + (size_t)i * D_INNER * DT_RANK, DT_RANK, D_INNER);
    wt_transpose_kernel<<<dim3(D_MODEL / 32, D_INNER / 32), 256, 0, stream>>>(
        ow  + (size_t)i * D_INNER * D_MODEL,
        owT + (size_t)i * D_MODEL * D_INNER, D_INNER, D_MODEL);
  }
  f32_to_bf16_kernel<<<(int)(MD / 4 + 255) / 256, 256, 0, stream>>>(x, xbf, (int)(MD / 4));

  for (int i = 0; i < N_LAYERS; ++i) {
    const float* xin = (i == 0) ? x : xbuf;
    const __bf16* WiT = inT + (size_t)i * D_MODEL * 2 * D_INNER;
    const __bf16* xpiT = xpT + (size_t)i * PROJ_W * D_INNER;
    const __bf16* dtiT = dtT + (size_t)i * D_INNER * DT_RANK;
    const __bf16* owiT = owT + (size_t)i * D_MODEL * D_INNER;
    const float* cwi = cw    + (size_t)i * D_INNER * D_CONV;
    const float* cbi = cb    + (size_t)i * D_INNER;
    const float* dbi = dtb   + (size_t)i * D_INNER;
    const float* Ai  = A_log + (size_t)i * D_INNER * D_STATE;
    const float* Di  = Dv    + (size_t)i * D_INNER;
    const float* lwi = lnw   + (size_t)i * D_MODEL;
    const float* lbi = lnb   + (size_t)i * D_MODEL;

    // 1. xz = x @ in_proj_w        (2048 x 4096 x 1024)
    wmma_gemm_bf16_kernel<<<dim3(M / 128, (2 * D_INNER) / 128), 256, 0, stream>>>(
        xbf, D_MODEL, WiT, D_MODEL, xz, 2 * D_INNER, nullptr,
        M, 2 * D_INNER, D_MODEL, nullptr, 0);

    // 2. depthwise causal conv + SiLU (fp32 + bf16 outputs)
    conv_silu_kernel<<<(B_SZ * SEQ * D_INNER) / 256, 256, 0, stream>>>(
        xz, cwi, cbi, ucv, ucvbf);

    // 3. proj = u @ x_proj_w       (2048 x 96 x 2048), bf16 copy for dt GEMM
    wmma_gemm_bf16_kernel<<<dim3(M / 128, (PROJ_W + 127) / 128), 256, 0, stream>>>(
        ucvbf, D_INNER, xpiT, D_INNER, proj, PROJ_W, projbf,
        M, PROJ_W, D_INNER, nullptr, 0);

    // 4. delta = softplus(dt @ dt_proj_w + dt_proj_b)  (A = projbf cols [0,64), lda=96)
    wmma_gemm_bf16_kernel<<<dim3(M / 128, D_INNER / 128), 256, 0, stream>>>(
        projbf, PROJ_W, dtiT, DT_RANK, delta, D_INNER, nullptr,
        M, D_INNER, DT_RANK, dbi, 1);

    // 5. selective scan + u*D + silu(z) gating -> bf16
    scan_kernel<<<B_SZ * (D_INNER / 256), 256, 0, stream>>>(
        delta, ucv, xz, proj, Ai, Di, ysbf);

    // 6. ymm = y @ out_proj_w      (2048 x 1024 x 2048)
    wmma_gemm_bf16_kernel<<<dim3(M / 128, D_MODEL / 128), 256, 0, stream>>>(
        ysbf, D_INNER, owiT, D_INNER, ymm, D_MODEL, nullptr,
        M, D_MODEL, D_INNER, nullptr, 0);

    // 7. x = LayerNorm(ymm + x); bf16 copy feeds next layer's in_proj GEMM
    float*  xout   = (i == N_LAYERS - 1) ? (float*)d_out : xbuf;
    __bf16* xoutbf = (i == N_LAYERS - 1) ? nullptr : xbf;
    residual_ln_kernel<<<M, 256, 0, stream>>>(ymm, xin, lwi, lbi, xout, xoutbf);
  }
}
